// CrossModalAttention_49606872269027
// MI455X (gfx1250) — matrix operand order repaired and checked by reading, not compiled, not run on hardware
//
#include <hip/hip_runtime.h>

#define BATCH 16
#define NHEAD 12
#define DMODEL 768
#define EPSV 1e-5f

typedef __attribute__((ext_vector_type(16))) __bf16 v16bf;
typedef __attribute__((ext_vector_type(8)))  float  v8f;
typedef __attribute__((ext_vector_type(4)))  unsigned v4u;
typedef __attribute__((ext_vector_type(8)))  unsigned v8u;

union FragB16 { v16bf v; unsigned u[8]; };

__device__ __forceinline__ unsigned short f2bf(float f) {
  union { float f; unsigned u; } x; x.f = f;
  unsigned r = x.u + 0x7FFFu + ((x.u >> 16) & 1u);   // RNE
  return (unsigned short)(r >> 16);
}

__device__ __forceinline__ unsigned pack_bf16(float a, float b) {
#if __has_builtin(__builtin_amdgcn_cvt_pk_bf16_f32)
  auto t = __builtin_amdgcn_cvt_pk_bf16_f32(a, b);
  unsigned u; __builtin_memcpy(&u, &t, 4);
  return u;
#else
  return (unsigned)f2bf(a) | ((unsigned)f2bf(b) << 16);
#endif
}

// ---------------------------------------------------------------------------
// TDM: 2D bf16 tile (tile_w x tile_h) -> LDS with 1-dword pad per row.
// pad_code: pad after 2^(pad_code+1) dwords (= tile_w*2 bytes).
// ---------------------------------------------------------------------------
__device__ __forceinline__ void tdm_load_2d(
    unsigned lds_addr, const void* gaddr, unsigned row_stride_elems,
    unsigned total_rows, unsigned tile_w, unsigned tile_h, unsigned pad_code)
{
  unsigned long long ga = (unsigned long long)(size_t)gaddr;
  v4u g0 = { 1u,                                   // count=1 (valid user D#)
             lds_addr,                             // lds_addr
             (unsigned)(ga & 0xFFFFFFFFu),         // global_addr[31:0]
             (unsigned)((ga >> 32) & 0x01FFFFFFu) | 0x80000000u };  // type=2
  v8u g1 = {
    (1u << 16) | (1u << 20) | (pad_code << 22),    // 2B elems, pad_en, interval, 1dw pad
    (row_stride_elems & 0xFFFFu) << 16,            // tensor_dim0[15:0]
    (row_stride_elems >> 16) | ((total_rows & 0xFFFFu) << 16),
    ((total_rows >> 16) & 0xFFFFu) | (tile_w << 16),
    tile_h,                                        // tile_dim1, tile_dim2=0
    row_stride_elems,                              // tensor_dim0_stride lo
    0u, 0u };
  asm volatile("tensor_load_to_lds %0, %1" :: "s"(g0), "s"(g1) : "memory");
}

// ---------------------------------------------------------------------------
// One-time conversions: f32 -> bf16 (features), f32 -> bf16 transposed (weights)
// ---------------------------------------------------------------------------
__global__ __launch_bounds__(256) void to_bf16_kernel(
    const float* __restrict__ x, unsigned* __restrict__ y, int n4)
{
  int i = blockIdx.x * 256 + threadIdx.x;
  if (i < n4) {
    const float4* xv = (const float4*)x;
    float4 v = xv[i];
    y[2 * i]     = pack_bf16(v.x, v.y);
    y[2 * i + 1] = pack_bf16(v.z, v.w);
  }
}

__global__ __launch_bounds__(256) void wtrans_bf16(
    const float* __restrict__ W, unsigned short* __restrict__ Wt)
{
  __shared__ float tile[32][33];
  const int tx = threadIdx.x & 31, g4 = (threadIdx.x >> 5) * 4;
  const int k0 = blockIdx.x * 32, n0 = blockIdx.y * 32;
  #pragma unroll
  for (int i = 0; i < 4; i++)
    tile[g4 + i][tx] = W[(size_t)(n0 + g4 + i) * DMODEL + k0 + tx];
  __syncthreads();
  #pragma unroll
  for (int i = 0; i < 4; i++)   // Wt[k][n] = W[n][k]
    Wt[(size_t)(k0 + g4 + i) * DMODEL + n0 + tx] = f2bf(tile[tx][g4 + i]);
}

// ---------------------------------------------------------------------------
// GEMM: out[M,768] = A[M,768] (bf16) @ Wt[768,768] (bf16, pre-transposed [k][n])
//       + bias. Block tile 128x128, 8 waves, wave tile 32x64 (2x4 WMMA tiles).
// Both LDS tiles streamed by the Tensor Data Mover, double-buffered.
// ---------------------------------------------------------------------------
__global__ __launch_bounds__(256) void gemm_bf16(
    const unsigned short* __restrict__ A, const unsigned short* __restrict__ Wt,
    const float* __restrict__ bias, void* __restrict__ outp,
    int M, int bf16out)
{
  __shared__ __align__(16) unsigned short As[2][128 * 34];  // [m][k], stride 34
  __shared__ __align__(16) unsigned short Bs[2][32 * 130];  // [k][n], stride 130

  const int tid  = threadIdx.x;
  const int lane = tid & 31;
  const int wid  = tid >> 5;
  const int wm   = wid & 3;
  const int wn   = wid >> 2;
  const int m0   = blockIdx.x * 128;
  const int n0   = blockIdx.y * 128;
  const int NIT  = DMODEL / 32;   // 24

  v8f acc[2][4];
  for (int i = 0; i < 2; i++) for (int j = 0; j < 4; j++) acc[i][j] = {};

  auto issue_tiles = [&](int k0, int buf) {
    // A tile: 32 wide x 128 rows, row pad -> stride 34 (pad after 16 dwords: code 3)
    tdm_load_2d((unsigned)(size_t)(void*)As[buf],
                A + (size_t)m0 * DMODEL + k0, DMODEL, (unsigned)M, 32u, 128u, 3u);
    // B tile: 128 wide x 32 rows, row pad -> stride 130 (pad after 64 dwords: code 5)
    tdm_load_2d((unsigned)(size_t)(void*)Bs[buf],
                Wt + (size_t)k0 * DMODEL + n0, DMODEL, (unsigned)DMODEL, 128u, 32u, 5u);
  };

  if (wid == 0) issue_tiles(0, 0);

  for (int it = 0; it < NIT; ++it) {
    if (wid == 0) __builtin_amdgcn_s_wait_tensorcnt(0);
    __syncthreads();
    if (wid == 0 && it + 1 < NIT) issue_tiles((it + 1) * 32, (it + 1) & 1);

    const unsigned* As32 = (const unsigned*)As[it & 1];
    const unsigned* Bs32 = (const unsigned*)Bs[it & 1];

    FragB16 a[2], b[4];
    const int kh = (lane < 16) ? 0 : 4;          // K-half select (dwords)
    #pragma unroll
    for (int sm = 0; sm < 2; sm++) {
      int m = wm * 32 + sm * 16 + (lane & 15);
      #pragma unroll
      for (int v = 0; v < 4; v++) {
        a[sm].u[v]     = As32[m * 17 + kh + v];
        a[sm].u[4 + v] = As32[m * 17 + 8 + kh + v];
      }
    }
    #pragma unroll
    for (int sn = 0; sn < 4; sn++) {
      int nh = (wn * 64 + sn * 16) >> 1;
      #pragma unroll
      for (int v = 0; v < 8; v++) { int kk = 2 * (kh + v + (v & 4)); int n = 2 * nh + (lane & 15); b[sn].u[v] = (unsigned)Bs[it & 1][kk * 130 + n] | ((unsigned)Bs[it & 1][(kk + 1) * 130 + n] << 16); }
    }
    #pragma unroll
    for (int sm = 0; sm < 2; sm++)
      #pragma unroll
      for (int sn = 0; sn < 4; sn++)
        acc[sm][sn] = __builtin_amdgcn_wmma_f32_16x16x32_bf16(
            false, a[sm].v, false, b[sn].v, (short)0, acc[sm][sn], false, false);
  }

  const int g = lane >> 4, nl = lane & 15;
  if (bf16out) {
    unsigned short* o16 = (unsigned short*)outp;
    #pragma unroll
    for (int sm = 0; sm < 2; sm++)
      #pragma unroll
      for (int sn = 0; sn < 4; sn++) {
        int ncol = n0 + wn * 64 + sn * 16 + nl;
        float bv = bias[ncol];
        #pragma unroll
        for (int r = 0; r < 8; r++) {
          int mrow = m0 + wm * 32 + sm * 16 + g * 8 + r;
          o16[(size_t)mrow * DMODEL + ncol] = f2bf(acc[sm][sn][r] + bv);
        }
      }
  } else {
    float* o32 = (float*)outp;
    #pragma unroll
    for (int sm = 0; sm < 2; sm++)
      #pragma unroll
      for (int sn = 0; sn < 4; sn++) {
        int ncol = n0 + wn * 64 + sn * 16 + nl;
        float bv = bias[ncol];
        #pragma unroll
        for (int r = 0; r < 8; r++) {
          int mrow = m0 + wm * 32 + sm * 16 + g * 8 + r;
          o32[(size_t)mrow * DMODEL + ncol] = acc[sm][sn][r] + bv;
        }
      }
  }
}

// ---------------------------------------------------------------------------
// Flash attention. bf16 Q/K/V merged-head [B*S, 768]; bf16 out (feeds GEMM).
// 8 waves x 16 queries. K transposed into LDS by VALU path; V tile via TDM.
// ---------------------------------------------------------------------------
__global__ __launch_bounds__(256) void flash_attn(
    const unsigned short* __restrict__ Qp, const unsigned short* __restrict__ Kp,
    const unsigned short* __restrict__ Vp, const unsigned char* __restrict__ maskp,
    unsigned short* __restrict__ O, int q_len, int k_len)
{
  __shared__ __align__(16) unsigned short KT[64 * 66];        // [hd][key]
  __shared__ __align__(16) unsigned short Vs[64 * 66];        // [key][hd], TDM pad
  __shared__ __align__(16) unsigned short Pscr[8][16 * 34];   // per-wave P tile

  const int tid = threadIdx.x, lane = tid & 31, wid = tid >> 5;
  const int b = blockIdx.y / NHEAD, h = blockIdx.y % NHEAD;
  const int qbase = blockIdx.x * 128 + wid * 16;

  const unsigned* Qd   = (const unsigned*)Qp;
  const unsigned* Kd   = (const unsigned*)Kp;
  const unsigned* KT32 = (const unsigned*)KT;
  const unsigned* Vs32 = (const unsigned*)Vs;
  const unsigned* P32  = (const unsigned*)Pscr[wid];

  FragB16 qf[2];
  {
    int m  = lane & 15;
    int kh = (lane < 16) ? 0 : 4;
    size_t rowbase = ((size_t)(b * q_len + qbase + m) * DMODEL + h * 64) >> 1;
    #pragma unroll
    for (int f = 0; f < 2; f++) {
      #pragma unroll
      for (int v = 0; v < 4; v++) {
        qf[f].u[v]     = Qd[rowbase + f * 16 + kh + v];
        qf[f].u[4 + v] = Qd[rowbase + f * 16 + 8 + kh + v];
      }
    }
  }

  v8f acc[4];
  for (int t = 0; t < 4; t++) acc[t] = {};
  float mrun[8], lrun[8];
  #pragma unroll
  for (int r = 0; r < 8; r++) { mrun[r] = -1e30f; lrun[r] = 0.f; }

  const float scale = 0.125f;  // 1/sqrt(64)

  for (int kc = 0; kc < k_len; kc += 64) {
    __syncthreads();

    if (wid == 0) {  // V tile [64 keys x 64 hd] via Tensor Data Mover
      tdm_load_2d((unsigned)(size_t)(void*)Vs,
                  Vp + ((size_t)(b * k_len + kc) * DMODEL + h * 64),
                  DMODEL, (unsigned)(BATCH * k_len), 64u, 64u, 4u);
      __builtin_amdgcn_s_wait_tensorcnt(0);
    }

    { // cooperative K chunk load + transpose into KT[hd][key]
      int r  = tid >> 2;
      int j0 = (tid & 3) * 8;
      size_t kr = ((size_t)(b * k_len + kc + r) * DMODEL + h * 64) >> 1;
      #pragma unroll
      for (int j = 0; j < 8; j++) {
        unsigned dk = Kd[kr + j0 + j];
        int c = 2 * (j0 + j);
        KT[c * 66 + r]       = (unsigned short)(dk & 0xFFFFu);
        KT[(c + 1) * 66 + r] = (unsigned short)(dk >> 16);
      }
      if (kc + 64 < k_len)
        __builtin_prefetch(Kd + kr + (size_t)64 * (DMODEL / 2), 0, 1);
    }
    __syncthreads();

    #pragma unroll
    for (int s = 0; s < 4; s++) {          // 16-key subtiles
      FragB16 kb0, kb1;
      #pragma unroll
      for (int v = 0; v < 8; v++) {
        int kk = 2 * (((lane < 16) ? 0 : 4) + v + (v & 4)), n = s * 16 + (lane & 15);
        kb0.u[v] = (unsigned)KT[kk * 66 + n] | ((unsigned)KT[(kk + 1) * 66 + n] << 16);
        kb1.u[v] = (unsigned)KT[(kk + 32) * 66 + n] | ((unsigned)KT[(kk + 33) * 66 + n] << 16);
      }
      v8f S = {};
      S = __builtin_amdgcn_wmma_f32_16x16x32_bf16(false, qf[0].v, false, kb0.v,
                                                  (short)0, S, false, false);
      S = __builtin_amdgcn_wmma_f32_16x16x32_bf16(false, qf[1].v, false, kb1.v,
                                                  (short)0, S, false, false);

      bool valid = true;
      if (maskp) {
        int keyg = kc + s * 16 + (lane & 15);
        valid = maskp[(size_t)b * k_len + keyg] != 0;
      }

      float p[8], alpha[8];
      #pragma unroll
      for (int r = 0; r < 8; r++) {
        float sv = valid ? S[r] * scale : -1e30f;
        float cm = sv;
        cm = fmaxf(cm, __shfl_xor(cm, 1, 32));
        cm = fmaxf(cm, __shfl_xor(cm, 2, 32));
        cm = fmaxf(cm, __shfl_xor(cm, 4, 32));
        cm = fmaxf(cm, __shfl_xor(cm, 8, 32));
        float mn = fmaxf(mrun[r], cm);
        alpha[r] = __expf(mrun[r] - mn);
        float pe = (sv < -1e29f) ? 0.f : __expf(sv - mn);
        float rs = pe;
        rs += __shfl_xor(rs, 1, 32);
        rs += __shfl_xor(rs, 2, 32);
        rs += __shfl_xor(rs, 4, 32);
        rs += __shfl_xor(rs, 8, 32);
        lrun[r] = lrun[r] * alpha[r] + rs;
        mrun[r] = mn;
        p[r] = pe;
      }
      #pragma unroll
      for (int t = 0; t < 4; t++)
        #pragma unroll
        for (int r = 0; r < 8; r++) acc[t][r] *= alpha[r];

      { // stash P tile (C-layout -> row-major bf16 in per-wave scratch)
        int gg = lane >> 4, nl = lane & 15, kkb = (s & 1) * 16;
        unsigned short* pp = Pscr[wid];
        #pragma unroll
        for (int r = 0; r < 8; r++) pp[(gg * 8 + r) * 34 + kkb + nl] = f2bf(p[r]);
      }

      if (s & 1) {  // every 32 keys: P x V
        asm volatile("s_wait_dscnt 0" ::: "memory");
        FragB16 pf;
        int m = lane & 15, kh = (lane < 16) ? 0 : 4;
        #pragma unroll
        for (int v = 0; v < 4; v++) {
          pf.u[v]     = P32[m * 17 + kh + v];
          pf.u[4 + v] = P32[m * 17 + 8 + kh + v];
        }
        int kb32 = (s >> 1) * 32;
        #pragma unroll
        for (int t = 0; t < 4; t++) {
          FragB16 vb;
          #pragma unroll
          for (int v = 0; v < 8; v++) {
            int kk = kb32 + 2 * (kh + v + (v & 4)), n = t * 16 + (lane & 15);
            vb.u[v] = (unsigned)Vs[kk * 66 + n] | ((unsigned)Vs[(kk + 1) * 66 + n] << 16);
          }
          acc[t] = __builtin_amdgcn_wmma_f32_16x16x32_bf16(
              false, pf.v, false, vb.v, (short)0, acc[t], false, false);
        }
      }
    }
  }

  const int gg = lane >> 4, nl = lane & 15;
  #pragma unroll
  for (int r = 0; r < 8; r++) {
    float inv = 1.f / fmaxf(lrun[r], 1e-20f);
    size_t base = (size_t)(b * q_len + qbase + gg * 8 + r) * DMODEL + h * 64;
    #pragma unroll
    for (int t = 0; t < 4; t++) O[base + t * 16 + nl] = f2bf(acc[t][r] * inv);
  }
}

// ---------------------------------------------------------------------------
// out = LayerNorm(proj + residual) * gamma + beta ; one block per row of 768
// ---------------------------------------------------------------------------
__global__ __launch_bounds__(256) void add_ln(
    const float* __restrict__ proj, const float* __restrict__ res,
    const float* __restrict__ gamma, const float* __restrict__ beta,
    float* __restrict__ out)
{
  __shared__ float red[256];
  const int tid = threadIdx.x;
  const size_t base = (size_t)blockIdx.x * DMODEL;

  float x0 = proj[base + tid]       + res[base + tid];
  float x1 = proj[base + tid + 256] + res[base + tid + 256];
  float x2 = proj[base + tid + 512] + res[base + tid + 512];

  red[tid] = x0 + x1 + x2;
  __syncthreads();
  for (int o = 128; o > 0; o >>= 1) { if (tid < o) red[tid] += red[tid + o]; __syncthreads(); }
  float mu = red[0] * (1.f / DMODEL);
  __syncthreads();

  red[tid] = x0 * x0 + x1 * x1 + x2 * x2;
  __syncthreads();
  for (int o = 128; o > 0; o >>= 1) { if (tid < o) red[tid] += red[tid + o]; __syncthreads(); }
  float var = red[0] * (1.f / DMODEL) - mu * mu;
  float inv = rsqrtf(var + EPSV);

  out[base + tid]       = (x0 - mu) * inv * gamma[tid]       + beta[tid];
  out[base + tid + 256] = (x1 - mu) * inv * gamma[tid + 256] + beta[tid + 256];
  out[base + tid + 512] = (x2 - mu) * inv * gamma[tid + 512] + beta[tid + 512];
}

// ---------------------------------------------------------------------------
extern "C" void kernel_launch(void* const* d_in, const int* in_sizes, int n_in,
                              void* d_out, int out_size, void* d_ws, size_t ws_size,
                              hipStream_t stream)
{
  (void)in_sizes; (void)n_in; (void)out_size; (void)ws_size;

  const float* intent  = (const float*)d_in[0];
  const float* context = (const float*)d_in[1];
  const unsigned char* mask = (const unsigned char*)d_in[2];
  const float* w_q  = (const float*)d_in[3],  *b_q  = (const float*)d_in[4];
  const float* w_k  = (const float*)d_in[5],  *b_k  = (const float*)d_in[6];
  const float* w_v  = (const float*)d_in[7],  *b_v  = (const float*)d_in[8];
  const float* w_qr = (const float*)d_in[9],  *b_qr = (const float*)d_in[10];
  const float* w_kr = (const float*)d_in[11], *b_kr = (const float*)d_in[12];
  const float* w_vr = (const float*)d_in[13], *b_vr = (const float*)d_in[14];
  const float* w_io = (const float*)d_in[15], *b_io = (const float*)d_in[16];
  const float* w_co = (const float*)d_in[17], *b_co = (const float*)d_in[18];
  const float* ln_i_w = (const float*)d_in[19], *ln_i_b = (const float*)d_in[20];
  const float* ln_c_w = (const float*)d_in[21], *ln_c_b = (const float*)d_in[22];

  const int Mq = BATCH * 512;    // 8192 intent rows
  const int Mk = BATCH * 1024;   // 16384 context rows

  char* ws = (char*)d_ws;
  size_t off = 0;
  auto salloc = [&](size_t bytes) {
    void* p = ws + off;
    off += (bytes + 255) & ~(size_t)255;
    return p;
  };

  unsigned short* intB = (unsigned short*)salloc((size_t)Mq * DMODEL * 2);
  unsigned short* ctxB = (unsigned short*)salloc((size_t)Mk * DMODEL * 2);
  const size_t WSZ = (size_t)DMODEL * DMODEL * 2;
  unsigned short* Wtq  = (unsigned short*)salloc(WSZ);
  unsigned short* Wtk  = (unsigned short*)salloc(WSZ);
  unsigned short* Wtv  = (unsigned short*)salloc(WSZ);
  unsigned short* Wtqr = (unsigned short*)salloc(WSZ);
  unsigned short* Wtkr = (unsigned short*)salloc(WSZ);
  unsigned short* Wtvr = (unsigned short*)salloc(WSZ);
  unsigned short* Wtio = (unsigned short*)salloc(WSZ);
  unsigned short* Wtco = (unsigned short*)salloc(WSZ);
  unsigned short* Qb  = (unsigned short*)salloc((size_t)Mq * DMODEL * 2);
  unsigned short* Kb  = (unsigned short*)salloc((size_t)Mk * DMODEL * 2);
  unsigned short* Vb  = (unsigned short*)salloc((size_t)Mk * DMODEL * 2);
  unsigned short* Qrb = (unsigned short*)salloc((size_t)Mk * DMODEL * 2);
  unsigned short* Krb = (unsigned short*)salloc((size_t)Mq * DMODEL * 2);
  unsigned short* Vrb = (unsigned short*)salloc((size_t)Mq * DMODEL * 2);
  unsigned short* attF = (unsigned short*)salloc((size_t)Mq * DMODEL * 2);
  unsigned short* attR = (unsigned short*)salloc((size_t)Mk * DMODEL * 2);
  float* projI = (float*)salloc((size_t)Mq * DMODEL * 4);
  float* projC = (float*)salloc((size_t)Mk * DMODEL * 4);

  dim3 blk(256);
  dim3 tgrid(DMODEL / 32, DMODEL / 32);   // 24 x 24

  // one-time precision conversion
  to_bf16_kernel<<<dim3((Mq * DMODEL / 4 + 255) / 256), blk, 0, stream>>>(intent,  (unsigned*)intB, Mq * DMODEL / 4);
  to_bf16_kernel<<<dim3((Mk * DMODEL / 4 + 255) / 256), blk, 0, stream>>>(context, (unsigned*)ctxB, Mk * DMODEL / 4);
  wtrans_bf16<<<tgrid, blk, 0, stream>>>(w_q,  Wtq);
  wtrans_bf16<<<tgrid, blk, 0, stream>>>(w_k,  Wtk);
  wtrans_bf16<<<tgrid, blk, 0, stream>>>(w_v,  Wtv);
  wtrans_bf16<<<tgrid, blk, 0, stream>>>(w_qr, Wtqr);
  wtrans_bf16<<<tgrid, blk, 0, stream>>>(w_kr, Wtkr);
  wtrans_bf16<<<tgrid, blk, 0, stream>>>(w_vr, Wtvr);
  wtrans_bf16<<<tgrid, blk, 0, stream>>>(w_io, Wtio);
  wtrans_bf16<<<tgrid, blk, 0, stream>>>(w_co, Wtco);

  // projections (TDM-fed bf16 GEMMs); N tiles = 768/128 = 6
  gemm_bf16<<<dim3(Mq / 128, 6), blk, 0, stream>>>(intB, Wtq,  b_q,  Qb,  Mq, 1);
  gemm_bf16<<<dim3(Mk / 128, 6), blk, 0, stream>>>(ctxB, Wtk,  b_k,  Kb,  Mk, 1);
  gemm_bf16<<<dim3(Mk / 128, 6), blk, 0, stream>>>(ctxB, Wtv,  b_v,  Vb,  Mk, 1);
  gemm_bf16<<<dim3(Mk / 128, 6), blk, 0, stream>>>(ctxB, Wtqr, b_qr, Qrb, Mk, 1);
  gemm_bf16<<<dim3(Mq / 128, 6), blk, 0, stream>>>(intB, Wtkr, b_kr, Krb, Mq, 1);
  gemm_bf16<<<dim3(Mq / 128, 6), blk, 0, stream>>>(intB, Wtvr, b_vr, Vrb, Mq, 1);

  // attention both directions
  flash_attn<<<dim3(512 / 128,  BATCH * NHEAD), blk, 0, stream>>>(Qb,  Kb,  Vb,  mask,    attF, 512, 1024);
  flash_attn<<<dim3(1024 / 128, BATCH * NHEAD), blk, 0, stream>>>(Qrb, Krb, Vrb, nullptr, attR, 1024, 512);

  // output projections (f32 out, pre-LN)
  gemm_bf16<<<dim3(Mq / 128, 6), blk, 0, stream>>>(attF, Wtio, b_io, projI, Mq, 0);
  gemm_bf16<<<dim3(Mk / 128, 6), blk, 0, stream>>>(attR, Wtco, b_co, projC, Mk, 0);

  // residual + LayerNorm into concatenated output
  float* outI = (float*)d_out;
  float* outC = outI + (size_t)Mq * DMODEL;
  add_ln<<<dim3(Mq), blk, 0, stream>>>(projI, intent,  ln_i_w, ln_i_b, outI);
  add_ln<<<dim3(Mk), blk, 0, stream>>>(projC, context, ln_c_w, ln_c_b, outC);
}
